// ScoreModel_78512002171213
// MI455X (gfx1250) — matrix operand order repaired and checked by reading, not compiled, not run on hardware
//
#include <hip/hip_runtime.h>
#include <hip/hip_bf16.h>
#include <math.h>

typedef __bf16 bf16;
typedef __attribute__((ext_vector_type(16))) __bf16 v16bf;
typedef __attribute__((ext_vector_type(8)))  __bf16 v8bf;
typedef __attribute__((ext_vector_type(8)))  float  v8f;
typedef __attribute__((ext_vector_type(4)))  unsigned int u32x4;
typedef __attribute__((ext_vector_type(4)))  int i32x4;
typedef __attribute__((ext_vector_type(8)))  int i32x8;

#define PI_F 3.14159265358979323846f

__device__ __forceinline__ float silu_f(float x) {
  return x / (1.0f + __expf(-x));
}

// ---------------- time embedding: temb = silu(silu(t@w1+b1)@w2+b2) ----------
__global__ void temb_kernel(const float* __restrict__ t,
                            const float* __restrict__ w1, const float* __restrict__ b1,
                            const float* __restrict__ w2, const float* __restrict__ b2,
                            float* __restrict__ temb) {
  __shared__ float h1[64];
  const int b = blockIdx.x, n = threadIdx.x;
  float acc = b1[n];
  for (int k = 0; k < 64; ++k) acc += t[b * 64 + k] * w1[k * 64 + n];
  h1[n] = silu_f(acc);
  __syncthreads();
  float a2 = b2[n];
  for (int k = 0; k < 64; ++k) a2 += h1[k] * w2[k * 64 + n];
  temb[b * 64 + n] = silu_f(a2);
}

// ---------------- node embedding ---------------------------------------------
__global__ void embed_kernel(const int* __restrict__ x, const int* __restrict__ batch,
                             const float* __restrict__ ae, const float* __restrict__ temb,
                             float* __restrict__ s, bf16* __restrict__ sbf, int N) {
  const int idx = blockIdx.x * blockDim.x + threadIdx.x;
  if (idx >= N * 64) return;
  const int n = idx >> 6, c = idx & 63;
  float val = ae[(x[n * 3 + 0]) * 64 + c]
            + ae[(16 + x[n * 3 + 1]) * 64 + c]
            + ae[(32 + x[n * 3 + 2]) * 64 + c]
            + temb[batch[n] * 64 + c];
  s[idx] = val;
  sbf[idx] = (bf16)val;
}

__global__ void zero_kernel(float* __restrict__ p, int n) {
  const int i = blockIdx.x * blockDim.x + threadIdx.x;
  if (i < n) p[i] = 0.0f;
}

// ---------------- edge geometry: rn, rbf||bond (bf16), degree count ----------
__global__ void edge_geom_kernel(const int* __restrict__ src, const int* __restrict__ tgt,
                                 const int* __restrict__ ea, const float* __restrict__ pos,
                                 const float* __restrict__ bond_emb, float cutoff,
                                 float* __restrict__ rn, bf16* __restrict__ ef,
                                 float* __restrict__ cnt, int E) {
  const int e = blockIdx.x * blockDim.x + threadIdx.x;
  if (e >= E) return;
  const int a = src[e], b = tgt[e];
  const float rx = pos[b * 3 + 0] - pos[a * 3 + 0];
  const float ry = pos[b * 3 + 1] - pos[a * 3 + 1];
  const float rz = pos[b * 3 + 2] - pos[a * 3 + 2];
  const float d = sqrtf(fmaxf(rx * rx + ry * ry + rz * rz, 1e-6f));
  rn[e * 3 + 0] = rx / d;
  rn[e * 3 + 1] = ry / d;
  rn[e * 3 + 2] = rz / d;
  const float env = (d < cutoff) ? 0.5f * (__cosf(PI_F * d / cutoff) + 1.0f) : 0.0f;
  const float w = cutoff / 16.0f;
  bf16* pe = ef + (size_t)e * 32;
#pragma unroll
  for (int j = 0; j < 16; ++j) {
    const float ctr = cutoff * (float)j / 15.0f;
    const float z = (d - ctr) / w;
    pe[j] = (bf16)(__expf(-0.5f * z * z) * env);
  }
  const int bt = ea[e];
#pragma unroll
  for (int j = 0; j < 16; ++j) pe[16 + j] = (bf16)bond_emb[bt * 16 + j];
  atomicAdd(&cnt[b], 1.0f);
}

__global__ void invcnt_kernel(const float* __restrict__ cnt, float* __restrict__ invc, int N) {
  const int i = blockIdx.x * blockDim.x + threadIdx.x;
  if (i < N) invc[i] = 1.0f / fmaxf(cnt[i], 1.0f);
}

// ---------------- weight prepack into wave32 B-fragment layout ---------------
// B layout (32x16 bf16): lanes 0-15 hold column N=lane, K=0..7,16..23 (2 per VGPR);
// lanes 16-31 hold same column, K=8..15,24..31.
__global__ void pack_wm_kernel(const float* __restrict__ wm, bf16* __restrict__ wpack, int total) {
  int idx = blockIdx.x * blockDim.x + threadIdx.x;
  if (idx >= total) return;
  const int t = idx & 15;
  int r = idx >> 4;
  const int lane = r & 31; r >>= 5;
  const int ks = r % 5; r /= 5;
  const int nt = r % 6; r /= 6;
  const int mat = r;
  const int n = (lane & 15) + 16 * nt;
  const int k = ks * 32 + (t & 7) + ((t >= 8) ? 16 : 0) + ((lane >= 16) ? 8 : 0);
  wpack[((size_t)mat * 30 + nt * 5 + ks) * 512 + lane * 16 + t] =
      (bf16)wm[((size_t)mat * 160 + k) * 96 + n];
}

__global__ void pack_wu_kernel(const float* __restrict__ wu, bf16* __restrict__ wpack, int total) {
  int idx = blockIdx.x * blockDim.x + threadIdx.x;
  if (idx >= total) return;
  const int t = idx & 15;
  int r = idx >> 4;
  const int lane = r & 31; r >>= 5;
  const int ks = r % 2; r /= 2;
  const int nt = r % 4; r /= 4;
  const int mat = r;
  const int n = (lane & 15) + 16 * nt;
  const int k = ks * 32 + (t & 7) + ((t >= 8) ? 16 : 0) + ((lane >= 16) ? 8 : 0);
  wpack[((size_t)mat * 8 + nt * 2 + ks) * 512 + lane * 16 + t] =
      (bf16)wu[((size_t)mat * 64 + k) * 64 + n];
}

// ---------------- edge MLP + message scatter (the hot kernel) ----------------
// One wave per 16-edge tile. feat = [s[src](64) | s[tgt](64) | rbf(16) | bond(16)]
// h = silu(feat @ wm + bm) via v_wmma_f32_16x16x32_bf16 (6 n-tiles x 5 k-steps).
// The 30KB packed weight block is staged into LDS once per workgroup with the
// Tensor Data Mover (tensor_load_to_lds, TENSORcnt), then read via ds_load.
#define WMP_DW 7680  // 30*512 bf16 = 7680 dwords

__global__ void __launch_bounds__(256) edge_mlp_kernel(
    const int* __restrict__ src, const int* __restrict__ tgt,
    const float* __restrict__ rn, const bf16* __restrict__ ef,
    const bf16* __restrict__ sbf, const float* __restrict__ vvec,
    const bf16* __restrict__ wpack, const float* __restrict__ bm,
    float* __restrict__ agg_s, float* __restrict__ agg_v, int E) {
  __shared__ bf16 lds_w[30 * 512];

  // ---- TDM: DMA packed weights (global -> LDS), issued by wave 0 ----
  if ((threadIdx.x >> 5) == 0) {
    const unsigned lds_off = (unsigned)(size_t)(&lds_w[0]);
    const unsigned long long ga = (unsigned long long)(size_t)wpack;
    u32x4 g0;
    g0[0] = 1u;                                   // count=1, user descriptor
    g0[1] = lds_off;                              // lds_addr (bytes)
    g0[2] = (unsigned)(ga & 0xffffffffu);         // global_addr[31:0]
    g0[3] = (unsigned)((ga >> 32) & 0x1ffffffu)   // global_addr[56:32]
          | (2u << 30);                           // type=2 (image)
    i32x8 g1;
    g1[0] = (int)(2u << 16);                      // data_size=4B; no multicast
    g1[1] = (int)((WMP_DW & 0xffffu) << 16);      // tensor_dim0[15:0]
    g1[2] = (int)(((WMP_DW >> 16) & 0xffffu) | (1u << 16)); // dim0[31:16], dim1=1
    g1[3] = (int)((WMP_DW & 0xffffu) << 16);      // tile_dim0 = 7680 dwords
    g1[4] = 0;                                    // tile_dim1=0 (1-D tile)
    g1[5] = (int)WMP_DW;                          // tensor_dim0_stride
    g1[6] = 0;
    g1[7] = 0;
    i32x4 z4 = {0, 0, 0, 0};
#if defined(__clang_major__) && __clang_major__ >= 23
    i32x8 z8 = {0, 0, 0, 0, 0, 0, 0, 0};
    __builtin_amdgcn_tensor_load_to_lds(g0, g1, z4, z4, z8, 0);
#else
    __builtin_amdgcn_tensor_load_to_lds(g0, g1, z4, z4, 0);
#endif
    __builtin_amdgcn_s_wait_tensorcnt(0);
  }
  __syncthreads();

  const int lane = threadIdx.x & 31;
  const int tile = blockIdx.x * (blockDim.x >> 5) + (threadIdx.x >> 5);
  const int e0 = tile * 16;
  if (e0 < E) {
    const int m16 = lane & 15;
    int eIdx = e0 + m16;
    if (eIdx >= E) eIdx = E - 1;  // clamp; contributions masked below
    const int sm = src[eIdx];
    const int tm = tgt[eIdx];
    const float rnx = rn[eIdx * 3 + 0];
    const float rny = rn[eIdx * 3 + 1];
    const float rnz = rn[eIdx * 3 + 2];
    const int kb = (lane >= 16) ? 8 : 0;  // 16-bit A layout lane split

    v16bf A[5];
    {
      const bf16* ps = sbf + (size_t)sm * 64;
      const bf16* pt = sbf + (size_t)tm * 64;
#pragma unroll
      for (int tp = 0; tp < 2; ++tp) {
        v8bf lo = *(const v8bf*)(ps + tp * 32 + kb);
        v8bf hi = *(const v8bf*)(ps + tp * 32 + kb + 16);
        v8bf lo2 = *(const v8bf*)(pt + tp * 32 + kb);
        v8bf hi2 = *(const v8bf*)(pt + tp * 32 + kb + 16);
#pragma unroll
        for (int q = 0; q < 8; ++q) {
          A[tp][q] = lo[q];      A[tp][q + 8] = hi[q];
          A[tp + 2][q] = lo2[q]; A[tp + 2][q + 8] = hi2[q];
        }
      }
      const bf16* pe = ef + (size_t)eIdx * 32;
      v8bf lo = *(const v8bf*)(pe + kb);
      v8bf hi = *(const v8bf*)(pe + kb + 16);
#pragma unroll
      for (int q = 0; q < 8; ++q) { A[4][q] = lo[q]; A[4][q + 8] = hi[q]; }
    }

    float h[6][8];
#pragma unroll
    for (int nt = 0; nt < 6; ++nt) {
      v8f acc = {};  // inline-0 C operand on the first WMMA
#pragma unroll
      for (int ks = 0; ks < 5; ++ks) {
        v16bf Bf = *(const v16bf*)(lds_w + ((nt * 5 + ks) * 32 + lane) * 16);
        acc = __builtin_amdgcn_wmma_f32_16x16x32_bf16(false, A[ks], false, Bf,
                                                      (short)0, acc, false, false);
      }
      const float bn = bm[m16 + 16 * nt];
#pragma unroll
      for (int r = 0; r < 8; ++r) h[nt][r] = silu_f(acc[r] + bn);
    }

    // scatter: ms -> agg_s[tgt], vm = v[src]*wv1 + rn (x) wv2 -> agg_v[tgt]
    const int mhi = (lane >= 16) ? 8 : 0;
#pragma unroll
    for (int r = 0; r < 8; ++r) {
      const int m = r + mhi;
      const int tEdge = __shfl(tm, m, 32);
      const int sEdge = __shfl(sm, m, 32);
      const float r0 = __shfl(rnx, m, 32);
      const float r1 = __shfl(rny, m, 32);
      const float r2 = __shfl(rnz, m, 32);
      if (e0 + m < E) {
#pragma unroll
        for (int nt = 0; nt < 4; ++nt)
          atomicAdd(&agg_s[(size_t)tEdge * 64 + m16 + 16 * nt], h[nt][r]);
        const float w1 = h[4][r], w2 = h[5][r];
        const float* vs = vvec + (size_t)sEdge * 48 + m16;
        float* vd = agg_v + (size_t)tEdge * 48 + m16;
        atomicAdd(vd + 0,  vs[0]  * w1 + r0 * w2);
        atomicAdd(vd + 16, vs[16] * w1 + r1 * w2);
        atomicAdd(vd + 32, vs[32] * w1 + r2 * w2);
      }
    }
  }
}

// ---------------- node update: s += silu(mean(agg_s) @ wu + bu) --------------
__global__ void __launch_bounds__(256) node_update_kernel(
    const float* __restrict__ agg_s, const float* __restrict__ invc,
    const bf16* __restrict__ wupack, const float* __restrict__ bu,
    float* __restrict__ s, bf16* __restrict__ sbf, int N) {
  const int lane = threadIdx.x & 31;
  const int tile = blockIdx.x * (blockDim.x >> 5) + (threadIdx.x >> 5);
  const int n0 = tile * 16;
  if (n0 >= N) return;
  const int m16 = lane & 15;
  const int node = n0 + m16;
  const float ic = invc[node];
  const int kb = (lane >= 16) ? 8 : 0;
  v16bf A[2];
  const float* pa = agg_s + (size_t)node * 64;
#pragma unroll
  for (int ks = 0; ks < 2; ++ks)
#pragma unroll
    for (int q = 0; q < 8; ++q) {
      A[ks][q]     = (bf16)(pa[ks * 32 + kb + q] * ic);
      A[ks][q + 8] = (bf16)(pa[ks * 32 + kb + 16 + q] * ic);
    }
  const int mhi = (lane >= 16) ? 8 : 0;
#pragma unroll
  for (int nt = 0; nt < 4; ++nt) {
    const int n = m16 + 16 * nt;
    v8f acc = {};
#pragma unroll
    for (int ks = 0; ks < 2; ++ks) {
      v16bf Bf = *(const v16bf*)(wupack + ((size_t)(nt * 2 + ks) * 32 + lane) * 16);
      acc = __builtin_amdgcn_wmma_f32_16x16x32_bf16(false, A[ks], false, Bf,
                                                    (short)0, acc, false, false);
    }
    const float bn = bu[n];
#pragma unroll
    for (int r = 0; r < 8; ++r) {
      const size_t p = (size_t)(n0 + r + mhi) * 64 + n;
      const float val = s[p] + silu_f(acc[r] + bn);
      s[p] = val;
      sbf[p] = (bf16)val;
    }
  }
}

__global__ void v_update_kernel(float* __restrict__ v, const float* __restrict__ agg_v,
                                const float* __restrict__ invc, int N) {
  const int i = blockIdx.x * blockDim.x + threadIdx.x;
  if (i >= N * 48) return;
  v[i] += agg_v[i] * invc[i / 48];
}

// ---------------- layernorm over channel dim (64) ----------------------------
__global__ void layernorm_kernel(float* __restrict__ s, bf16* __restrict__ sbf,
                                 const float* __restrict__ g, const float* __restrict__ b) {
  __shared__ float red[64];
  const int n = blockIdx.x, c = threadIdx.x;
  const size_t p = (size_t)n * 64 + c;
  const float x = s[p];
  red[c] = x;
  __syncthreads();
  for (int off = 32; off > 0; off >>= 1) { if (c < off) red[c] += red[c + off]; __syncthreads(); }
  const float mean = red[0] * (1.0f / 64.0f);
  __syncthreads();
  const float d = x - mean;
  red[c] = d * d;
  __syncthreads();
  for (int off = 32; off > 0; off >>= 1) { if (c < off) red[c] += red[c + off]; __syncthreads(); }
  const float var = red[0] * (1.0f / 64.0f);
  const float y = d * rsqrtf(var + 1e-5f) * g[c] + b[c];
  s[p] = y;
  sbf[p] = (bf16)y;
}

// ---------------- head -------------------------------------------------------
__global__ void dsc_kernel(const float* __restrict__ s, const float* __restrict__ w,
                           float* __restrict__ dsc) {
  __shared__ float red[64];
  const int n = blockIdx.x, c = threadIdx.x;
  red[c] = silu_f(s[(size_t)n * 64 + c]) * w[c];
  __syncthreads();
  for (int off = 32; off > 0; off >>= 1) { if (c < off) red[c] += red[c + off]; __syncthreads(); }
  if (c == 0) dsc[n] = red[0];
}

__global__ void sr_kernel(const int* __restrict__ src, const int* __restrict__ tgt,
                          const float* __restrict__ pos, const float* __restrict__ dsc,
                          float* __restrict__ sr, int E) {
  const int e = blockIdx.x * blockDim.x + threadIdx.x;
  if (e >= E) return;
  const int a = src[e], b = tgt[e];
  const float val = dsc[a] + dsc[b];
  atomicAdd(&sr[b * 3 + 0], val * (pos[a * 3 + 0] - pos[b * 3 + 0]));
  atomicAdd(&sr[b * 3 + 1], val * (pos[a * 3 + 1] - pos[b * 3 + 1]));
  atomicAdd(&sr[b * 3 + 2], val * (pos[a * 3 + 2] - pos[b * 3 + 2]));
}

__global__ void out_kernel(const float* __restrict__ v, const float* __restrict__ how,
                           const float* __restrict__ sr, float* __restrict__ out, int N) {
  const int idx = blockIdx.x * blockDim.x + threadIdx.x;
  if (idx >= N * 3) return;
  const int n = idx / 3, c = idx % 3;
  float acc = sr[idx];
#pragma unroll
  for (int j = 0; j < 16; ++j) acc += v[(size_t)n * 48 + c * 16 + j] * how[j];
  out[idx] = acc;
}

// -----------------------------------------------------------------------------
extern "C" void kernel_launch(void* const* d_in, const int* in_sizes, int n_in,
                              void* d_out, int out_size, void* d_ws, size_t ws_size,
                              hipStream_t stream) {
  (void)n_in; (void)out_size; (void)ws_size;
  const int*   x     = (const int*)d_in[0];
  const float* t     = (const float*)d_in[1];
  const float* pos   = (const float*)d_in[2];
  const int*   eil   = (const int*)d_in[3];
  const int*   eig   = (const int*)d_in[4];
  const int*   eal   = (const int*)d_in[5];
  const int*   eag   = (const int*)d_in[6];
  const int*   batch = (const int*)d_in[7];
  const float* ae    = (const float*)d_in[8];
  const float* bond  = (const float*)d_in[9];
  const float* tw1   = (const float*)d_in[10];
  const float* tb1   = (const float*)d_in[11];
  const float* tw2   = (const float*)d_in[12];
  const float* tb2   = (const float*)d_in[13];
  const float* wm    = (const float*)d_in[14];
  const float* bm    = (const float*)d_in[15];
  const float* wu    = (const float*)d_in[16];
  const float* bu    = (const float*)d_in[17];
  const float* lng   = (const float*)d_in[18];
  const float* lnb   = (const float*)d_in[19];
  const float* hdw   = (const float*)d_in[20];
  const float* how   = (const float*)d_in[21];
  float* out = (float*)d_out;

  const int N  = in_sizes[0] / 3;
  const int B  = in_sizes[1] / 64;
  const int EL = in_sizes[3] / 2;
  const int EG = in_sizes[4] / 2;

  char* base = (char*)d_ws;
  size_t off = 0;
  auto alloc = [&](size_t bytes) -> void* {
    void* p = base + off;
    off += (bytes + 255) & ~(size_t)255;
    return p;
  };
  float* temb   = (float*)alloc((size_t)B * 64 * 4);
  float* s      = (float*)alloc((size_t)N * 64 * 4);
  bf16*  sbf    = (bf16*)alloc((size_t)N * 64 * 2);
  float* v      = (float*)alloc((size_t)N * 48 * 4);
  float* agg_s  = (float*)alloc((size_t)N * 64 * 4);
  float* agg_v  = (float*)alloc((size_t)N * 48 * 4);
  float* rn_l   = (float*)alloc((size_t)EL * 3 * 4);
  float* rn_g   = (float*)alloc((size_t)EG * 3 * 4);
  bf16*  ef_l   = (bf16*)alloc((size_t)EL * 32 * 2);
  bf16*  ef_g   = (bf16*)alloc((size_t)EG * 32 * 2);
  float* cnt_l  = (float*)alloc((size_t)N * 4);
  float* cnt_g  = (float*)alloc((size_t)N * 4);
  float* invc_l = (float*)alloc((size_t)N * 4);
  float* invc_g = (float*)alloc((size_t)N * 4);
  bf16*  wmpack = (bf16*)alloc((size_t)10 * 30 * 512 * 2);
  bf16*  wupack = (bf16*)alloc((size_t)10 * 8 * 512 * 2);
  float* dscb   = (float*)alloc((size_t)N * 4);
  float* srb    = (float*)alloc((size_t)N * 3 * 4);

  const int T = 256;
  temb_kernel<<<B, 64, 0, stream>>>(t, tw1, tb1, tw2, tb2, temb);
  embed_kernel<<<(N * 64 + T - 1) / T, T, 0, stream>>>(x, batch, ae, temb, s, sbf, N);
  zero_kernel<<<(N * 48 + T - 1) / T, T, 0, stream>>>(v, N * 48);
  zero_kernel<<<(N + T - 1) / T, T, 0, stream>>>(cnt_l, N);
  zero_kernel<<<(N + T - 1) / T, T, 0, stream>>>(cnt_g, N);
  zero_kernel<<<(N * 3 + T - 1) / T, T, 0, stream>>>(srb, N * 3);
  edge_geom_kernel<<<(EL + T - 1) / T, T, 0, stream>>>(eil, eil + EL, eal, pos, bond, 3.0f,
                                                       rn_l, ef_l, cnt_l, EL);
  edge_geom_kernel<<<(EG + T - 1) / T, T, 0, stream>>>(eig, eig + EG, eag, pos, bond, 10.0f,
                                                       rn_g, ef_g, cnt_g, EG);
  invcnt_kernel<<<(N + T - 1) / T, T, 0, stream>>>(cnt_l, invc_l, N);
  invcnt_kernel<<<(N + T - 1) / T, T, 0, stream>>>(cnt_g, invc_g, N);
  pack_wm_kernel<<<(153600 + T - 1) / T, T, 0, stream>>>(wm, wmpack, 153600);
  pack_wu_kernel<<<(40960 + T - 1) / T, T, 0, stream>>>(wu, wupack, 40960);

  for (int i = 0; i < 5; ++i) {
    for (int k = 0; k < 2; ++k) {
      const int    E   = k ? EG : EL;
      const int*   sp  = k ? eig : eil;
      const int*   tp  = sp + E;
      const float* rnp = k ? rn_g : rn_l;
      const bf16*  efp = k ? ef_g : ef_l;
      const float* iv  = k ? invc_g : invc_l;
      const int mat = i * 2 + k;
      zero_kernel<<<(N * 64 + T - 1) / T, T, 0, stream>>>(agg_s, N * 64);
      zero_kernel<<<(N * 48 + T - 1) / T, T, 0, stream>>>(agg_v, N * 48);
      const int nTiles = (E + 15) / 16;
      edge_mlp_kernel<<<(nTiles + 7) / 8, 256, 0, stream>>>(
          sp, tp, rnp, efp, sbf, v, wmpack + (size_t)mat * 30 * 512, bm + mat * 96,
          agg_s, agg_v, E);
      const int nTilesN = N / 16;
      node_update_kernel<<<(nTilesN + 7) / 8, 256, 0, stream>>>(
          agg_s, iv, wupack + (size_t)mat * 8 * 512, bu + mat * 64, s, sbf, N);
      v_update_kernel<<<(N * 48 + T - 1) / T, T, 0, stream>>>(v, agg_v, iv, N);
    }
    layernorm_kernel<<<N, 64, 0, stream>>>(s, sbf, lng + i * 64, lnb + i * 64);
  }

  dsc_kernel<<<N, 64, 0, stream>>>(s, hdw, dscb);
  sr_kernel<<<(EG + T - 1) / T, T, 0, stream>>>(eig, eig + EG, pos, dscb, srb, EG);
  out_kernel<<<(N * 3 + T - 1) / T, T, 0, stream>>>(v, how, srb, out, N);
}
